// VAELSTM_5334349382111
// MI455X (gfx1250) — compile-verified, matrix operand order checked by reading, not run on hardware
//
#include <hip/hip_runtime.h>
#include <math.h>

// ------------------------------------------------------------------
// CDNA5 (gfx1250) VAE+LSTM pipeline.
// All GEMM-shaped work (convs via implicit GEMM, linear layers, LSTM
// gates) goes through v_wmma_f32_16x16x32_bf16 with a bf16x3 split of
// the fp32 operands (aH*bH + aH*bL + aL*bH, f32 accumulate) to keep
// near-fp32 accuracy at bf16-WMMA throughput.
// Round 3: GEMM templated on B-operand mode so each instantiation's
// K-loop has a single branch-free gather path (no uniform-mode
// branches inside the hot loop); branchless clamped loads, hoisted
// column decode, float4 A-fragment loads, dual accumulators.
// ------------------------------------------------------------------

typedef __attribute__((ext_vector_type(16))) __bf16 v16bf;
typedef __attribute__((ext_vector_type(8)))  float  v8f;

struct GemmP {
  const float* A; int lda;          // A: M x K row-major (weights)
  const float* B;                   // plain matrix or activation tensor
  int M, N, K;
  int bMode;                        // 0 plain, 1 conv(k3,pad1), 2 convT(lhs_dil2,pad(1,2))
  int bLdK, bColStride, bColOff;    // plain: B[k][c] = B[k*bLdK + c*bColStride + bColOff]
  int Cin, Hin, Win, Hout, Wout, stride; // conv modes, activations channel-major [Cin][Nimg*Hin*Win]
  int inHW, outHW;
  long long actChanStride;          // Nimg*Hin*Win
  const float* bias1;               // per-row (m) bias, may be null
  const float* bias2;               // second per-row bias, may be null
  const float* addSrc; int addLdM, addLdN, addOff;  // optional residual add
  float* out; long long outLdM; int outLdN, outOff; // out[m*outLdM + c*outLdN + outOff]
  int accumulate;                   // 1: out += result
  int act;                          // 0 none, 1 exact GELU, 2 tanh
};

// One 16x16 C tile per wave; 8 waves/block => 64(M) x 32(N) block tile.
// MODE: 0 plain affine B, 1 implicit-GEMM conv, 2 implicit-GEMM convT.
template <int MODE>
__global__ __launch_bounds__(256) void gemm_wmma(GemmP p) {
  const int lane = threadIdx.x & 31;
  const int wave = threadIdx.x >> 5;
  const int tileM = blockIdx.y * 64 + (wave & 3) * 16;
  const int tileN = blockIdx.x * 32 + (wave >> 2) * 16;
  const int kh  = lane >> 4;   // K-half select per ISA A/B fragment layout
  const int l15 = lane & 15;
  const int mRow = tileM + l15;
  const int nCol = tileN + l15;

  const int K = p.K;
  // Clamp OOB rows/cols to 0: they only pollute C rows/cols we never store,
  // so loads stay safe and branch-free with no masking needed.
  const int mr = (mRow < p.M) ? mRow : 0;
  const float* aBase = p.A + (long long)mr * p.lda;

  const int nc = (nCol < p.N) ? nCol : 0;
  const int Hin = p.Hin, Win = p.Win;
  const int bLdK = p.bLdK;
  const long long chanStride = p.actChanStride;
  const float* bPlain = nullptr;
  const float* actBase = nullptr;
  int iy0 = 0, ix0 = 0;
  if (MODE == 0) {
    bPlain = p.B + (long long)nc * p.bColStride + p.bColOff;
  } else {
    // hoist column decode: n -> (image, oy, ox); loop-invariant per lane
    int nImg = nc / p.outHW;
    int rem  = nc - nImg * p.outHW;
    int oy = rem / p.Wout;
    int ox = rem - oy * p.Wout;
    actBase = p.B + (long long)nImg * p.inHW;
    if (MODE == 1) { iy0 = oy * p.stride - 1; ix0 = ox * p.stride - 1; }
    else           { iy0 = oy - 1;            ix0 = ox - 1;            }
  }

  // branch-free B gather for a valid k (address-select + value-select)
  auto loadB_el = [&](int k) -> float {
    if (MODE == 0) {
      return bPlain[(long long)k * bLdK];
    }
    int ci = k / 9;
    int r9 = k - ci * 9;
    int ky = r9 / 3;
    int kx = r9 - ky * 3;
    int iy = iy0 + ky, ix = ix0 + kx;
    bool ok;
    if (MODE == 1) {
      ok = ((unsigned)iy < (unsigned)Hin) & ((unsigned)ix < (unsigned)Win);
    } else {  // transposed conv: dilated coords must be even and in range
      ok = ((unsigned)iy <= (unsigned)(2 * Hin - 2)) &
           ((unsigned)ix <= (unsigned)(2 * Win - 2)) &
           (((iy | ix) & 1) == 0);
      iy >>= 1; ix >>= 1;
    }
    long long off = (long long)ci * chanStride + iy * Win + ix;
    off = ok ? off : 0;            // clamp to safe address
    float v = actBase[off];
    return ok ? v : 0.0f;
  };

  v8f accHH = {}, accX = {};       // dual accumulators: hi*hi and cross terms
  for (int kb = 0; kb < K; kb += 32) {
    float aF[16], bF[16];
    if (kb + 32 <= K) {
      // full chunk: A fragment = two contiguous 8-float runs -> 4x float4
      const float4* ap0 = reinterpret_cast<const float4*>(aBase + kb + 8 * kh);
      const float4* ap1 = reinterpret_cast<const float4*>(aBase + kb + 16 + 8 * kh);
      float4 a0 = ap0[0], a1 = ap0[1], a2 = ap1[0], a3 = ap1[1];
      aF[0]=a0.x; aF[1]=a0.y; aF[2]=a0.z;  aF[3]=a0.w;
      aF[4]=a1.x; aF[5]=a1.y; aF[6]=a1.z;  aF[7]=a1.w;
      aF[8]=a2.x; aF[9]=a2.y; aF[10]=a2.z; aF[11]=a2.w;
      aF[12]=a3.x; aF[13]=a3.y; aF[14]=a3.z; aF[15]=a3.w;
#pragma unroll
      for (int e = 0; e < 16; ++e) {
        int k = kb + ((e < 8) ? (8 * kh + e) : (16 + 8 * kh + (e - 8)));
        bF[e] = loadB_el(k);
      }
    } else {
      // K tail: clamp k and zero-select (only path that needs k masking)
#pragma unroll
      for (int e = 0; e < 16; ++e) {
        int k = kb + ((e < 8) ? (8 * kh + e) : (16 + 8 * kh + (e - 8)));
        int kc = (k < K) ? k : 0;
        float av = aBase[kc];
        float bv = loadB_el(kc);
        bool kin = (k < K);
        aF[e] = kin ? av : 0.0f;
        bF[e] = kin ? bv : 0.0f;
      }
    }
    v16bf ahi{}, alo{}, bhi{}, blo{};
#pragma unroll
    for (int e = 0; e < 16; ++e) {
      __bf16 ah = (__bf16)aF[e];
      __bf16 bh = (__bf16)bF[e];
      ahi[e] = ah; bhi[e] = bh;
      alo[e] = (__bf16)(aF[e] - (float)ah);
      blo[e] = (__bf16)(bF[e] - (float)bh);
    }
    // bf16x3: aH*bH into accHH (independent), cross terms into accX
    accHH = __builtin_amdgcn_wmma_f32_16x16x32_bf16(false, ahi, false, bhi, (short)0, accHH, false, false);
    accX  = __builtin_amdgcn_wmma_f32_16x16x32_bf16(false, ahi, false, blo, (short)0, accX,  false, false);
    accX  = __builtin_amdgcn_wmma_f32_16x16x32_bf16(false, alo, false, bhi, (short)0, accX,  false, false);
  }

  if (nCol < p.N) {
#pragma unroll
    for (int r = 0; r < 8; ++r) {
      int m = tileM + r + 8 * kh;   // C layout: vgpr r, lanes 0-15 -> M=r, 16-31 -> M=8+r
      if (m < p.M) {
        float v = accHH[r] + accX[r];
        if (p.bias1) v += p.bias1[m];
        if (p.bias2) v += p.bias2[m];
        if (p.addSrc)
          v += p.addSrc[(long long)m * p.addLdM + (long long)nCol * p.addLdN + p.addOff];
        long long oidx = (long long)m * p.outLdM + (long long)nCol * p.outLdN + p.outOff;
        if (p.accumulate) v += p.out[oidx];
        if (p.act == 1)      v = 0.5f * v * (1.0f + erff(v * 0.7071067811865475f));
        else if (p.act == 2) v = tanhf(v);
        p.out[oidx] = v;
      }
    }
  }
}

// ---------------- small helper kernels ----------------

__global__ void zero_kernel(float* p, int n) {
  int i = blockIdx.x * blockDim.x + threadIdx.x;
  if (i < n) p[i] = 0.0f;
}

// encoder flatten: enc[c][n*16+sp] -> out[k=c*16+sp][n]   (64ch, 3200 imgs, 16 spatial)
__global__ void repack_flatten(const float* __restrict__ enc, float* __restrict__ out) {
  int i = blockIdx.x * blockDim.x + threadIdx.x;
  if (i >= 1024 * 3200) return;
  int kRow = i / 3200;
  int n    = i - kRow * 3200;
  int c = kRow >> 4, sp = kRow & 15;
  out[i] = enc[c * (3200 * 16) + n * 16 + sp];
}

// decoder reshape: hlin[k=c*16+sp][n] -> hdec[c][n*16+sp]  (64ch, 3168 imgs, 16 spatial)
__global__ void repack_split16(const float* __restrict__ hlin, float* __restrict__ out) {
  int i = blockIdx.x * blockDim.x + threadIdx.x;
  if (i >= 64 * 3168 * 16) return;
  int c = i / (3168 * 16);
  int rem = i - c * (3168 * 16);
  int n = rem >> 4, sp = rem & 15;
  out[i] = hlin[(c * 16 + sp) * 3168 + n];
}

__device__ __forceinline__ float sigmoidf(float x) { return 1.0f / (1.0f + expf(-x)); }

// LSTM cell: g [2048][32] feature-major, h/c [512][32]
__global__ void lstm_cell_kernel(const float* __restrict__ g, float* h, float* c) {
  int i = blockIdx.x * blockDim.x + threadIdx.x;
  if (i >= 512 * 32) return;
  int j = i >> 5, b = i & 31;
  float ig = g[j * 32 + b];
  float fg = g[(512 + j) * 32 + b];
  float gg = g[(1024 + j) * 32 + b];
  float og = g[(1536 + j) * 32 + b];
  float c2 = sigmoidf(fg) * c[i] + sigmoidf(ig) * tanhf(gg);
  c[i] = c2;
  h[i] = sigmoidf(og) * tanhf(c2);
}

__device__ __forceinline__ float rng_normal(unsigned long long idx) {
  unsigned long long s = idx * 0x9E3779B97F4A7C15ull + 0xD1B54A32D192ED03ull;
  s ^= s >> 30; s *= 0xBF58476D1CE4E5B9ull;
  s ^= s >> 27; s *= 0x94D049BB133111EBull;
  s ^= s >> 31;
  unsigned u1 = (unsigned)s, u2 = (unsigned)(s >> 32);
  float f1 = ((float)u1 + 0.5f) * (1.0f / 4294967296.0f);
  float f2 = ((float)u2) * (1.0f / 4294967296.0f);
  return sqrtf(-2.0f * logf(f1)) * cosf(6.283185307179586f * f2);
}

// z[j][col] = pred[j][col] + eps * exp(0.5*pred[64+j][col]),  col = b*99+t
__global__ void sample_kernel(const float* __restrict__ pred, float* __restrict__ z) {
  int i = blockIdx.x * blockDim.x + threadIdx.x;
  if (i >= 64 * 3168) return;
  int j = i / 3168, col = i - j * 3168;
  float mu = pred[j * 3168 + col];
  float lv = pred[(64 + j) * 3168 + col];
  float eps = rng_normal((unsigned long long)(col * 64 + j));
  z[i] = mu + eps * expf(0.5f * lv);
}

// loss = mean over b(32), t(3..98), j(128) of (pred - my)^2
__global__ void loss_reduce(const float* __restrict__ pred, const float* __restrict__ ml,
                            float* acc) {
  __shared__ float sdata[256];
  int tid = threadIdx.x;
  int i = blockIdx.x * blockDim.x + tid;
  float v = 0.0f;
  const int total = 32 * 96 * 128;
  if (i < total) {
    int b = i / (96 * 128);
    int rem = i - b * 96 * 128;
    int t3 = rem / 128;
    int j = rem - t3 * 128;
    int t = t3 + 3;
    float pv = pred[j * 3168 + b * 99 + t];
    float mv = ml[j * 3200 + b * 100 + (t + 1)];
    float d = pv - mv;
    v = d * d;
  }
  sdata[tid] = v;
  __syncthreads();
  for (int s = 128; s > 0; s >>= 1) {
    if (tid < s) sdata[tid] += sdata[tid + s];
    __syncthreads();
  }
  if (tid == 0) atomicAdd(acc, sdata[0]);
}

__global__ void loss_final(const float* acc, float* out) {
  if (threadIdx.x == 0 && blockIdx.x == 0)
    out[0] = acc[0] * (1.0f / (32.0f * 96.0f * 128.0f));
}

// ---------------- host orchestration ----------------

static inline void launch_gemm(hipStream_t st, const GemmP& p) {
  dim3 grid((unsigned)((p.N + 31) / 32), (unsigned)((p.M + 63) / 64));
  if (p.bMode == 0)      gemm_wmma<0><<<grid, dim3(256), 0, st>>>(p);
  else if (p.bMode == 1) gemm_wmma<1><<<grid, dim3(256), 0, st>>>(p);
  else                   gemm_wmma<2><<<grid, dim3(256), 0, st>>>(p);
}

extern "C" void kernel_launch(void* const* d_in, const int* in_sizes, int n_in,
                              void* d_out, int out_size, void* d_ws, size_t ws_size,
                              hipStream_t stream) {
  (void)in_sizes; (void)n_in; (void)out_size; (void)ws_size;
  const float* x      = (const float*)d_in[0];
  const float* enc_w1 = (const float*)d_in[2];  const float* enc_b1 = (const float*)d_in[3];
  const float* enc_w2 = (const float*)d_in[4];  const float* enc_b2 = (const float*)d_in[5];
  const float* enc_w3 = (const float*)d_in[6];  const float* enc_b3 = (const float*)d_in[7];
  const float* enc_w4 = (const float*)d_in[8];  const float* enc_b4 = (const float*)d_in[9];
  const float* enc_w5 = (const float*)d_in[10]; const float* enc_b5 = (const float*)d_in[11];
  const float* w_mu   = (const float*)d_in[12]; const float* b_mu   = (const float*)d_in[13];
  const float* w_lv   = (const float*)d_in[14]; const float* b_lv   = (const float*)d_in[15];
  const float* dec_wl = (const float*)d_in[16]; const float* dec_bl = (const float*)d_in[17];
  const float* dect_w1= (const float*)d_in[18]; const float* dect_b1= (const float*)d_in[19];
  const float* dec_w2 = (const float*)d_in[20]; const float* dec_b2 = (const float*)d_in[21];
  const float* dect_w3= (const float*)d_in[22]; const float* dect_b3= (const float*)d_in[23];
  const float* dec_w4 = (const float*)d_in[24]; const float* dec_b4 = (const float*)d_in[25];
  const float* dect_w5= (const float*)d_in[26]; const float* dect_b5= (const float*)d_in[27];
  const float* Wih0   = (const float*)d_in[28]; const float* Whh0   = (const float*)d_in[29];
  const float* bih0   = (const float*)d_in[30]; const float* bhh0   = (const float*)d_in[31];
  const float* WihR   = (const float*)d_in[32]; const float* WhhR   = (const float*)d_in[33];
  const float* bihR   = (const float*)d_in[34]; const float* bhhR   = (const float*)d_in[35];
  const float* Wout   = (const float*)d_in[36]; const float* bout   = (const float*)d_in[37];

  float* outF = (float*)d_out;   // [0]=loss, [1..] = y (3168*1024)

  // workspace layout (floats), ~205 MB total
  float* ws    = (float*)d_ws;
  float* bufA  = ws;                        // 26,214,400
  float* bufB  = bufA + 26214400;           // 26,214,400
  float* ml    = bufB + 26214400;           // 128*3200
  float* pred  = ml   + 128 * 3200;         // 128*3168
  float* zbuf  = pred + 128 * 3168;         // 64*3168
  float* gbuf  = zbuf + 64 * 3168;          // 2048*32
  float* hbuf  = gbuf + 2048 * 32;          // 5*512*32
  float* cbuf  = hbuf + 5 * 512 * 32;       // 5*512*32
  float* lossA = cbuf + 5 * 512 * 32;       // 1

  const int NIMG = 3200;   // 32*100 encoder images
  const int NDEC = 3168;   // 32*99 decoder samples

  GemmP p;

  // ---------------- Encoder ----------------
  // conv1: 1->32, 32x32 -> 16x16, stride 2
  p = {}; p.A = enc_w1; p.lda = 9; p.M = 32; p.K = 9; p.N = NIMG * 256;
  p.B = x; p.bMode = 1; p.Cin = 1; p.Hin = 32; p.Win = 32; p.Hout = 16; p.Wout = 16;
  p.stride = 2; p.inHW = 1024; p.outHW = 256; p.actChanStride = (long long)NIMG * 1024;
  p.bias1 = enc_b1; p.out = bufA; p.outLdM = NIMG * 256; p.outLdN = 1; p.act = 1;
  launch_gemm(stream, p);
  // conv2: 32->32, 16x16, stride 1
  p = {}; p.A = enc_w2; p.lda = 288; p.M = 32; p.K = 288; p.N = NIMG * 256;
  p.B = bufA; p.bMode = 1; p.Cin = 32; p.Hin = 16; p.Win = 16; p.Hout = 16; p.Wout = 16;
  p.stride = 1; p.inHW = 256; p.outHW = 256; p.actChanStride = (long long)NIMG * 256;
  p.bias1 = enc_b2; p.out = bufB; p.outLdM = NIMG * 256; p.outLdN = 1; p.act = 1;
  launch_gemm(stream, p);
  // conv3: 32->64, 16 -> 8, stride 2
  p = {}; p.A = enc_w3; p.lda = 288; p.M = 64; p.K = 288; p.N = NIMG * 64;
  p.B = bufB; p.bMode = 1; p.Cin = 32; p.Hin = 16; p.Win = 16; p.Hout = 8; p.Wout = 8;
  p.stride = 2; p.inHW = 256; p.outHW = 64; p.actChanStride = (long long)NIMG * 256;
  p.bias1 = enc_b3; p.out = bufA; p.outLdM = NIMG * 64; p.outLdN = 1; p.act = 1;
  launch_gemm(stream, p);
  // conv4: 64->64, 8x8, stride 1
  p = {}; p.A = enc_w4; p.lda = 576; p.M = 64; p.K = 576; p.N = NIMG * 64;
  p.B = bufA; p.bMode = 1; p.Cin = 64; p.Hin = 8; p.Win = 8; p.Hout = 8; p.Wout = 8;
  p.stride = 1; p.inHW = 64; p.outHW = 64; p.actChanStride = (long long)NIMG * 64;
  p.bias1 = enc_b4; p.out = bufB; p.outLdM = NIMG * 64; p.outLdN = 1; p.act = 1;
  launch_gemm(stream, p);
  // conv5: 64->64, 8 -> 4, stride 2
  p = {}; p.A = enc_w5; p.lda = 576; p.M = 64; p.K = 576; p.N = NIMG * 16;
  p.B = bufB; p.bMode = 1; p.Cin = 64; p.Hin = 8; p.Win = 8; p.Hout = 4; p.Wout = 4;
  p.stride = 2; p.inHW = 64; p.outHW = 16; p.actChanStride = (long long)NIMG * 64;
  p.bias1 = enc_b5; p.out = bufA; p.outLdM = NIMG * 16; p.outLdN = 1; p.act = 1;
  launch_gemm(stream, p);
  // flatten -> Bmu [1024][3200] in bufB
  repack_flatten<<<(1024 * 3200 + 255) / 256, 256, 0, stream>>>(bufA, bufB);
  // mu / lv heads -> ml [128][3200] feature-major
  p = {}; p.A = w_mu; p.lda = 1024; p.M = 64; p.K = 1024; p.N = NIMG;
  p.B = bufB; p.bMode = 0; p.bLdK = NIMG; p.bColStride = 1;
  p.bias1 = b_mu; p.out = ml; p.outLdM = NIMG; p.outLdN = 1;
  launch_gemm(stream, p);
  p.A = w_lv; p.bias1 = b_lv; p.out = ml + 64 * NIMG;
  launch_gemm(stream, p);

  // ---------------- LSTM ----------------
  {  // zero h, c, lossAcc (contiguous)
    int cnt = 5 * 512 * 32 * 2 + 1;
    zero_kernel<<<(cnt + 255) / 256, 256, 0, stream>>>(hbuf, cnt);
  }
  for (int t = 0; t < 99; ++t) {
    // layer 0: g = Wih0 * x_t  (+bih0+bhh0)
    p = {}; p.A = Wih0; p.lda = 128; p.M = 2048; p.K = 128; p.N = 32;
    p.B = ml; p.bMode = 0; p.bLdK = 3200; p.bColStride = 100; p.bColOff = t;
    p.bias1 = bih0; p.bias2 = bhh0; p.out = gbuf; p.outLdM = 32; p.outLdN = 1;
    launch_gemm(stream, p);
    // g += Whh0 * h0
    p = {}; p.A = Whh0; p.lda = 512; p.M = 2048; p.K = 512; p.N = 32;
    p.B = hbuf; p.bMode = 0; p.bLdK = 32; p.bColStride = 1;
    p.out = gbuf; p.outLdM = 32; p.outLdN = 1; p.accumulate = 1;
    launch_gemm(stream, p);
    lstm_cell_kernel<<<64, 256, 0, stream>>>(gbuf, hbuf, cbuf);
    // layers 1..4
    for (int i = 0; i < 4; ++i) {
      int l = i + 1;
      p = {}; p.A = WihR + (long long)i * 2048 * 512; p.lda = 512;
      p.M = 2048; p.K = 512; p.N = 32;
      p.B = hbuf + (l - 1) * 512 * 32; p.bMode = 0; p.bLdK = 32; p.bColStride = 1;
      p.bias1 = bihR + i * 2048; p.bias2 = bhhR + i * 2048;
      p.out = gbuf; p.outLdM = 32; p.outLdN = 1;
      launch_gemm(stream, p);
      p = {}; p.A = WhhR + (long long)i * 2048 * 512; p.lda = 512;
      p.M = 2048; p.K = 512; p.N = 32;
      p.B = hbuf + l * 512 * 32; p.bMode = 0; p.bLdK = 32; p.bColStride = 1;
      p.out = gbuf; p.outLdM = 32; p.outLdN = 1; p.accumulate = 1;
      launch_gemm(stream, p);
      lstm_cell_kernel<<<64, 256, 0, stream>>>(gbuf, hbuf + l * 512 * 32, cbuf + l * 512 * 32);
    }
    // pred_t = mx_t + Wout * h4 + bout   -> pred[j][b*99+t]
    p = {}; p.A = Wout; p.lda = 512; p.M = 128; p.K = 512; p.N = 32;
    p.B = hbuf + 4 * 512 * 32; p.bMode = 0; p.bLdK = 32; p.bColStride = 1;
    p.bias1 = bout;
    p.addSrc = ml; p.addLdM = 3200; p.addLdN = 100; p.addOff = t;
    p.out = pred; p.outLdM = 3168; p.outLdN = 99; p.outOff = t;
    launch_gemm(stream, p);
  }

  // loss
  loss_reduce<<<(32 * 96 * 128 + 255) / 256, 256, 0, stream>>>(pred, ml, lossA);
  loss_final<<<1, 64, 0, stream>>>(lossA, outF);

  // sample z
  sample_kernel<<<(64 * 3168 + 255) / 256, 256, 0, stream>>>(pred, zbuf);

  // ---------------- Decoder ----------------
  // linear 64->1024, gelu -> hlin [1024][3168] in bufB
  p = {}; p.A = dec_wl; p.lda = 64; p.M = 1024; p.K = 64; p.N = NDEC;
  p.B = zbuf; p.bMode = 0; p.bLdK = NDEC; p.bColStride = 1;
  p.bias1 = dec_bl; p.out = bufB; p.outLdM = NDEC; p.outLdN = 1; p.act = 1;
  launch_gemm(stream, p);
  // reshape -> channel-major [64][3168*16] in bufA
  repack_split16<<<(64 * 3168 * 16 + 255) / 256, 256, 0, stream>>>(bufB, bufA);
  // convT1: 64->64, 4 -> 8
  p = {}; p.A = dect_w1; p.lda = 576; p.M = 64; p.K = 576; p.N = NDEC * 64;
  p.B = bufA; p.bMode = 2; p.Cin = 64; p.Hin = 4; p.Win = 4; p.Hout = 8; p.Wout = 8;
  p.inHW = 16; p.outHW = 64; p.actChanStride = (long long)NDEC * 16;
  p.bias1 = dect_b1; p.out = bufB; p.outLdM = NDEC * 64; p.outLdN = 1; p.act = 1;
  launch_gemm(stream, p);
  // conv2: 64->64, 8x8, stride 1
  p = {}; p.A = dec_w2; p.lda = 576; p.M = 64; p.K = 576; p.N = NDEC * 64;
  p.B = bufB; p.bMode = 1; p.Cin = 64; p.Hin = 8; p.Win = 8; p.Hout = 8; p.Wout = 8;
  p.stride = 1; p.inHW = 64; p.outHW = 64; p.actChanStride = (long long)NDEC * 64;
  p.bias1 = dec_b2; p.out = bufA; p.outLdM = NDEC * 64; p.outLdN = 1; p.act = 1;
  launch_gemm(stream, p);
  // convT3: 64->32, 8 -> 16
  p = {}; p.A = dect_w3; p.lda = 576; p.M = 32; p.K = 576; p.N = NDEC * 256;
  p.B = bufA; p.bMode = 2; p.Cin = 64; p.Hin = 8; p.Win = 8; p.Hout = 16; p.Wout = 16;
  p.inHW = 64; p.outHW = 256; p.actChanStride = (long long)NDEC * 64;
  p.bias1 = dect_b3; p.out = bufB; p.outLdM = NDEC * 256; p.outLdN = 1; p.act = 1;
  launch_gemm(stream, p);
  // conv4: 32->32, 16x16, stride 1
  p = {}; p.A = dec_w4; p.lda = 288; p.M = 32; p.K = 288; p.N = NDEC * 256;
  p.B = bufB; p.bMode = 1; p.Cin = 32; p.Hin = 16; p.Win = 16; p.Hout = 16; p.Wout = 16;
  p.stride = 1; p.inHW = 256; p.outHW = 256; p.actChanStride = (long long)NDEC * 256;
  p.bias1 = dec_b4; p.out = bufA; p.outLdM = NDEC * 256; p.outLdN = 1; p.act = 1;
  launch_gemm(stream, p);
  // convT5: 32->1, 16 -> 32, tanh; write y straight into d_out+1
  p = {}; p.A = dect_w5; p.lda = 288; p.M = 1; p.K = 288; p.N = NDEC * 1024;
  p.B = bufA; p.bMode = 2; p.Cin = 32; p.Hin = 16; p.Win = 16; p.Hout = 32; p.Wout = 32;
  p.inHW = 256; p.outHW = 1024; p.actChanStride = (long long)NDEC * 256;
  p.bias1 = dect_b5; p.out = outF + 1; p.outLdM = (long long)NDEC * 1024; p.outLdN = 1; p.act = 2;
  launch_gemm(stream, p);
}
